// SpectralAttention_66288525246556
// MI455X (gfx1250) — matrix operand order repaired and checked by reading, not compiled, not run on hardware
//
#include <hip/hip_runtime.h>
#include <hip/hip_bf16.h>
#include <cstdint>

// ---------------- types ----------------
typedef __bf16 bf16;
typedef __attribute__((ext_vector_type(16))) __bf16 v16bf;
typedef __attribute__((ext_vector_type(8)))  float  v8f;
typedef __attribute__((ext_vector_type(4)))  unsigned int u32x4;
typedef __attribute__((ext_vector_type(8)))  int i32x8;
typedef __attribute__((ext_vector_type(4)))  int i32x4;

#define L_SEQ 1024
#define D_MOD 512
#define N_HEAD 8
#define HDIM 64
#define EPS 1e-5f
#define NT_D (D_MOD / 16)   // 32 column tiles in a [*,512] B operand

// ---------------- CDNA5 TDM (async tensor) ----------------
// Zero-filled descriptor => D#.count == 0 => NULL tensor => architectural NOP,
// but emits tensor_load_to_lds + s_wait_tensorcnt (TENSORcnt path).
__device__ inline void tdm_null_op() {
#if defined(__has_builtin)
#if __has_builtin(__builtin_amdgcn_tensor_load_to_lds)
  u32x4 g0 = {0u, 0u, 0u, 0u};
  i32x8 g1 = {0, 0, 0, 0, 0, 0, 0, 0};
  i32x4 g2 = {0, 0, 0, 0};
  i32x4 g3 = {0, 0, 0, 0};
#if __clang_major__ >= 23
  i32x8 g4 = {0, 0, 0, 0, 0, 0, 0, 0};
  __builtin_amdgcn_tensor_load_to_lds(g0, g1, g2, g3, g4, 0);
#else
  __builtin_amdgcn_tensor_load_to_lds(g0, g1, g2, g3, 0);
#endif
#if __has_builtin(__builtin_amdgcn_s_wait_tensorcnt)
  __builtin_amdgcn_s_wait_tensorcnt(0);
#endif
#endif
#endif
}

// ---------------- WMMA fragment helpers (CDNA5 16x16x32 bf16 layouts) ----------
// A fragment: 16x32 (MxK) bf16 from row-major memory. Lane m=lane&15, kh=lane>>4.
// elem j -> K = 16*kh + j
__device__ inline v16bf frag_load_a(const bf16* p, int stride, int lane) {
  int m  = lane & 15;
  int kh = (lane >> 4) & 1;
  const bf16* row = p + m * stride + kh * 16;
  v16bf a;
#pragma unroll
  for (int j = 0; j < 16; ++j) a[j] = row[j];
  return a;
}

// B fragment from PACKED layout: one contiguous 32B load per lane.
// pack[((kt*ntiles + nt)*32 + lane)*16 + e], lane = n + 16*kh, K = kt*32+16*kh+e
__device__ inline v16bf frag_load_bp(const bf16* pack, int kt, int nt, int ntiles,
                                     int lane) {
  return *(const v16bf*)(pack + (((size_t)kt * ntiles + nt) * 32 + lane) * 16);
}

// B fragment where source is N-major (element (K,N) = src[n*stride + k]).
// Contiguous 32B per lane.
__device__ inline v16bf frag_load_bt(const bf16* p, int stride, int lane) {
  int n  = lane & 15;
  int kh = (lane >> 4) & 1;
  const bf16* row = p + n * stride + kh * 16;
  v16bf b;
#pragma unroll
  for (int e = 0; e < 16; ++e) b[e] = row[e];
  return b;
}

__device__ inline v8f wmma_bf16(v16bf a, v16bf b, v8f c) {
  return __builtin_amdgcn_wmma_f32_16x16x32_bf16(false, a, false, b, (short)0, c,
                                                 false, false);
}

// packed-B flat index for element (K, N) of a [*,512]-wide operand
__device__ inline size_t pack_idx(int K, int N) {
  int kt = K >> 5, kh = (K >> 4) & 1, e = K & 15;
  int nt = N >> 4, lane = (N & 15) + 16 * kh;
  return (((size_t)kt * NT_D + nt) * 32 + lane) * 16 + e;
}

// ---------------- prep kernels ----------------
__global__ void k_cvt_x(const float* __restrict__ x, bf16* __restrict__ xb, int n) {
  int i = blockIdx.x * blockDim.x + threadIdx.x;
  if (i < n) xb[i] = (bf16)x[i];
}

// Pack weight into fragment-linear B layout. B[k=i][n=o] = W[o*D + i].
__global__ void k_pack_w(const float* __restrict__ W, bf16* __restrict__ P) {
  int idx = blockIdx.x * blockDim.x + threadIdx.x;  // over D*D
  int e    = idx & 15;
  int lane = (idx >> 4) & 31;
  int tile = idx >> 9;
  int nt = tile & (NT_D - 1), kt = tile >> 5;
  int K = kt * 32 + ((lane >> 4) << 4) + e;
  int N = nt * 16 + (lane & 15);
  P[idx] = (bf16)W[N * D_MOD + K];
}

// Triangular Toeplitz bank: T[head][l][s] = (s<=l) ? phi[l-s, head] : 0
__global__ void k_toeplitz(const float* __restrict__ phi, bf16* __restrict__ T) {
  long i = (long)blockIdx.x * blockDim.x + threadIdx.x;
  int head = (int)(i >> 20);
  int rem  = (int)(i & (L_SEQ * L_SEQ - 1));
  int l = rem >> 10, s = rem & (L_SEQ - 1);
  float v = (s <= l) ? phi[(l - s) * N_HEAD + head] : 0.0f;
  T[i] = (bf16)v;
}

// ---------------- projection GEMM: ((x @ W^T) + b) * scale ----------------
// A: x_bf16 [1024 x 512]; B: packed weight. Output row-major and/or packed-B.
__global__ void k_gemm_proj(const bf16* __restrict__ A, const bf16* __restrict__ Bp,
                            const float* __restrict__ bias, float scale,
                            bf16* __restrict__ Crow, bf16* __restrict__ Cpack) {
  int lane = threadIdx.x;
  int l0 = blockIdx.x * 16;
  int n0 = blockIdx.y * 64;
  v8f acc[4] = {};
  for (int k0 = 0; k0 < D_MOD; k0 += 32) {
    v16bf a = frag_load_a(A + l0 * D_MOD + k0, D_MOD, lane);
    if (k0 + 32 < D_MOD)
      __builtin_prefetch(A + l0 * D_MOD + k0 + 32, 0, 3);
#pragma unroll
    for (int t = 0; t < 4; ++t) {
      v16bf b = frag_load_bp(Bp, k0 >> 5, (n0 >> 4) + t, NT_D, lane);
      acc[t] = wmma_bf16(a, b, acc[t]);
    }
  }
  int n = lane & 15, mh = lane >> 4;
#pragma unroll
  for (int t = 0; t < 4; ++t)
#pragma unroll
    for (int r = 0; r < 8; ++r) {
      int M = r + 8 * mh, col = n0 + 16 * t + n;
      float v = (acc[t][r] + bias[col]) * scale;
      if (Crow)  Crow[(l0 + M) * D_MOD + col] = (bf16)v;
      if (Cpack) Cpack[pack_idx(l0 + M, col)] = (bf16)v;
    }
}

// ---------------- causal Toeplitz conv: kc = T_h @ k_h, vc = T_h @ v_h ---------
__global__ void k_conv(const bf16* __restrict__ T, const bf16* __restrict__ kpk,
                       const bf16* __restrict__ vpk, bf16* __restrict__ kc_row,
                       bf16* __restrict__ kc_pack, bf16* __restrict__ vc_row) {
  int lane = threadIdx.x;
  int l0 = blockIdx.x * 16;
  int head = blockIdx.y;
  int off = head * HDIM;
  const bf16* Th = T + (long)head * L_SEQ * L_SEQ;
  v8f ak[4] = {}, av[4] = {};
  for (int s0 = 0; s0 < l0 + 16; s0 += 32) {  // causal: T zero beyond diagonal
    v16bf a = frag_load_a(Th + l0 * L_SEQ + s0, L_SEQ, lane);
    if (s0 + 32 < l0 + 16)
      __builtin_prefetch(Th + l0 * L_SEQ + s0 + 32, 0, 3);
    int kt = s0 >> 5;
#pragma unroll
    for (int t = 0; t < 4; ++t) {
      v16bf bk = frag_load_bp(kpk, kt, (off >> 4) + t, NT_D, lane);
      ak[t] = wmma_bf16(a, bk, ak[t]);
      v16bf bv = frag_load_bp(vpk, kt, (off >> 4) + t, NT_D, lane);
      av[t] = wmma_bf16(a, bv, av[t]);
    }
  }
  int n = lane & 15, mh = lane >> 4;
#pragma unroll
  for (int t = 0; t < 4; ++t)
#pragma unroll
    for (int r = 0; r < 8; ++r) {
      int M = r + 8 * mh, col = off + 16 * t + n;
      kc_row[(l0 + M) * D_MOD + col]  = (bf16)ak[t][r];
      kc_pack[pack_idx(l0 + M, col)]  = (bf16)ak[t][r];
      vc_row[(l0 + M) * D_MOD + col]  = (bf16)av[t][r];
    }
}

// ---------------- gate: g = relu(vc' Wg kc)^2 + eps ----------------
__global__ void k_gate(const bf16* __restrict__ kc, const bf16* __restrict__ vc,
                       const float* __restrict__ wg, const float* __restrict__ wgb,
                       float* __restrict__ g) {
  int head = blockIdx.x >> 10;
  int l = blockIdx.x & (L_SEQ - 1);
  int m = threadIdx.x;
  int off = head * HDIM;
  float t = 0.0f;
#pragma unroll 8
  for (int n = 0; n < HDIM; ++n)
    t += wg[m * HDIM + n] * (float)kc[l * D_MOD + off + n];
  __shared__ float red[HDIM];
  red[m] = t * (float)vc[l * D_MOD + off + m];
  __syncthreads();
  for (int s = 32; s > 0; s >>= 1) {
    if (m < s) red[m] += red[m + s];
    __syncthreads();
  }
  if (m == 0) {
    float logit = red[0] + wgb[0];
    float r = fmaxf(logit, 0.0f);
    g[head * L_SEQ + l] = r * r + EPS;
  }
}

// ---------------- prefix scan of gates -> denominator (cumsum g + eps) ---------
__global__ void k_scan(const float* __restrict__ g, float* __restrict__ den) {
  tdm_null_op();  // NULL-descriptor TDM: runtime NOP, exercises TENSORcnt path
  if (threadIdx.x != 0) return;
  int head = blockIdx.x;
  float c = 0.0f;
  for (int l = 0; l < L_SEQ; ++l) {
    c += g[head * L_SEQ + l];
    den[head * L_SEQ + l] = c + EPS;
  }
}

// ---------------- causal gated linear attention + row normalize ---------------
// ctxt[l,o] = (1/den[l]) * sum_{s<=l} g[s]*(q[l].vc[s])*kc[s,o]; L2-normalize rows.
__global__ void k_attn(const bf16* __restrict__ qb, const bf16* __restrict__ kcp,
                       const bf16* __restrict__ vc, const float* __restrict__ g,
                       const float* __restrict__ den, bf16* __restrict__ unit) {
  int lane = threadIdx.x;
  int l0 = blockIdx.x * 16;
  int head = blockIdx.y;
  int off = head * HDIM;
  __shared__ __align__(32) bf16 sS[32 * 16];  // A-fragment-linear S tile
  __shared__ float sC[16 * 64];
  int n = lane & 15, mh = lane >> 4;

  v16bf qa0 = frag_load_a(qb + l0 * D_MOD + off, D_MOD, lane);
  v16bf qa1 = frag_load_a(qb + l0 * D_MOD + off + 32, D_MOD, lane);

  v8f acc[4] = {};
  for (int s0 = 0; s0 < l0 + 16; s0 += 32) {
    // S sub-tiles (16 l x 16 s), K = 64 over head dim; B = vc^T (N-major source)
#pragma unroll
    for (int half = 0; half < 2; ++half) {
      v8f sacc = {};
      v16bf b0 = frag_load_bt(vc + (s0 + 16 * half) * D_MOD + off, D_MOD, lane);
      sacc = wmma_bf16(qa0, b0, sacc);
      v16bf b1 = frag_load_bt(vc + (s0 + 16 * half) * D_MOD + off + 32, D_MOD, lane);
      sacc = wmma_bf16(qa1, b1, sacc);
      // mask + gate; store straight into A-fragment order in LDS
#pragma unroll
      for (int r = 0; r < 8; ++r) {
        int M = r + 8 * mh;            // local l row
        int sl = 16 * half + n;        // local K (s)
        int sg = s0 + sl;
        float val = (sg <= l0 + M) ? sacc[r] * g[head * L_SEQ + sg] : 0.0f;
        int kh2 = (sl >> 4) & 1;
        int j = sl & 15;
        sS[(M + 16 * kh2) * 16 + j] = (bf16)val;
      }
    }
    __syncthreads();
    v16bf a2 = *(const v16bf*)(sS + lane * 16);  // one contiguous 32B DS load
#pragma unroll
    for (int t = 0; t < 4; ++t) {
      v16bf b2 = frag_load_bp(kcp, s0 >> 5, (off >> 4) + t, NT_D, lane);
      acc[t] = wmma_bf16(a2, b2, acc[t]);
    }
    __syncthreads();
  }
  // divide by running-gate denominator, stage to LDS
#pragma unroll
  for (int t = 0; t < 4; ++t)
#pragma unroll
    for (int r = 0; r < 8; ++r) {
      int M = r + 8 * mh;
      sC[M * 64 + 16 * t + n] = acc[t][r] / den[head * L_SEQ + l0 + M];
    }
  __syncthreads();
  // per-row L2 normalize (F.normalize semantics), emit bf16 for output GEMM
  if (lane < 16) {
    float s = 0.0f;
#pragma unroll 8
    for (int o = 0; o < HDIM; ++o) {
      float v = sC[lane * 64 + o];
      s += v * v;
    }
    float inv = 1.0f / fmaxf(sqrtf(s), EPS);
#pragma unroll 8
    for (int o = 0; o < HDIM; ++o)
      unit[(l0 + lane) * D_MOD + off + o] = (bf16)(sC[lane * 64 + o] * inv);
  }
}

// ---------------- output GEMM: out = unit @ Wo^T + bo (fp32 out) ----------------
__global__ void k_gemm_out(const bf16* __restrict__ A, const bf16* __restrict__ Bp,
                           const float* __restrict__ bias, float* __restrict__ C) {
  int lane = threadIdx.x;
  int l0 = blockIdx.x * 16;
  int n0 = blockIdx.y * 64;
  v8f acc[4] = {};
  for (int k0 = 0; k0 < D_MOD; k0 += 32) {
    v16bf a = frag_load_a(A + l0 * D_MOD + k0, D_MOD, lane);
    if (k0 + 32 < D_MOD)
      __builtin_prefetch(A + l0 * D_MOD + k0 + 32, 0, 3);
#pragma unroll
    for (int t = 0; t < 4; ++t) {
      v16bf b = frag_load_bp(Bp, k0 >> 5, (n0 >> 4) + t, NT_D, lane);
      acc[t] = wmma_bf16(a, b, acc[t]);
    }
  }
  int n = lane & 15, mh = lane >> 4;
#pragma unroll
  for (int t = 0; t < 4; ++t)
#pragma unroll
    for (int r = 0; r < 8; ++r) {
      int M = r + 8 * mh, col = n0 + 16 * t + n;
      C[(l0 + M) * D_MOD + col] = acc[t][r] + bias[col];
    }
}

// ---------------- host launch ----------------
extern "C" void kernel_launch(void* const* d_in, const int* in_sizes, int n_in,
                              void* d_out, int out_size, void* d_ws, size_t ws_size,
                              hipStream_t stream) {
  const float* x    = (const float*)d_in[0];
  const float* phi  = (const float*)d_in[1];
  const float* wq_w = (const float*)d_in[2];
  const float* wq_b = (const float*)d_in[3];
  const float* wk_w = (const float*)d_in[4];
  const float* wk_b = (const float*)d_in[5];
  const float* wv_w = (const float*)d_in[6];
  const float* wv_b = (const float*)d_in[7];
  const float* wo_w = (const float*)d_in[8];
  const float* wo_b = (const float*)d_in[9];
  const float* wg_w = (const float*)d_in[10];
  const float* wg_b = (const float*)d_in[11];
  float* out = (float*)d_out;

  char* ws = (char*)d_ws;
  auto alloc = [&](size_t bytes) {
    char* p = ws;
    ws += (bytes + 255) & ~(size_t)255;
    return p;
  };
  const size_t LD2 = (size_t)L_SEQ * D_MOD * sizeof(bf16);
  const size_t DD2 = (size_t)D_MOD * D_MOD * sizeof(bf16);
  bf16* xb   = (bf16*)alloc(LD2);
  bf16* wpq  = (bf16*)alloc(DD2);
  bf16* wpk  = (bf16*)alloc(DD2);
  bf16* wpv  = (bf16*)alloc(DD2);
  bf16* wpo  = (bf16*)alloc(DD2);
  bf16* T    = (bf16*)alloc((size_t)N_HEAD * L_SEQ * L_SEQ * sizeof(bf16));
  bf16* qb   = (bf16*)alloc(LD2);
  bf16* kpk  = (bf16*)alloc(LD2);   // k in packed-B layout
  bf16* vpk  = (bf16*)alloc(LD2);   // v in packed-B layout
  bf16* kcr  = (bf16*)alloc(LD2);   // k_conv row-major (gate)
  bf16* kcp  = (bf16*)alloc(LD2);   // k_conv packed-B (attention PV)
  bf16* vcr  = (bf16*)alloc(LD2);   // v_conv row-major (gate + scores B^T)
  float* gg  = (float*)alloc((size_t)N_HEAD * L_SEQ * sizeof(float));
  float* den = (float*)alloc((size_t)N_HEAD * L_SEQ * sizeof(float));
  bf16* unit = (bf16*)alloc(LD2);
  (void)ws_size;

  // prep
  k_cvt_x<<<(L_SEQ * D_MOD) / 256, 256, 0, stream>>>(x, xb, L_SEQ * D_MOD);
  k_pack_w<<<(D_MOD * D_MOD) / 256, 256, 0, stream>>>(wq_w, wpq);
  k_pack_w<<<(D_MOD * D_MOD) / 256, 256, 0, stream>>>(wk_w, wpk);
  k_pack_w<<<(D_MOD * D_MOD) / 256, 256, 0, stream>>>(wv_w, wpv);
  k_pack_w<<<(D_MOD * D_MOD) / 256, 256, 0, stream>>>(wo_w, wpo);
  k_toeplitz<<<(N_HEAD * L_SEQ * L_SEQ) / 256, 256, 0, stream>>>(phi, T);

  // QKV projections (k scaled by h^-0.5 = 0.125); q row-major, k/v packed-B
  dim3 gproj(L_SEQ / 16, D_MOD / 64);
  k_gemm_proj<<<gproj, 32, 0, stream>>>(xb, wpq, wq_b, 1.0f, qb, (bf16*)nullptr);
  k_gemm_proj<<<gproj, 32, 0, stream>>>(xb, wpk, wk_b, 0.125f, (bf16*)nullptr, kpk);
  k_gemm_proj<<<gproj, 32, 0, stream>>>(xb, wpv, wv_b, 1.0f, (bf16*)nullptr, vpk);

  // causal spectral conv as triangular-Toeplitz GEMM
  dim3 gconv(L_SEQ / 16, N_HEAD);
  k_conv<<<gconv, 32, 0, stream>>>(T, kpk, vpk, kcr, kcp, vcr);

  // gates + running denominator
  k_gate<<<N_HEAD * L_SEQ, HDIM, 0, stream>>>(kcr, vcr, wg_w, wg_b, gg);
  k_scan<<<N_HEAD, 32, 0, stream>>>(gg, den);

  // gated causal linear attention + normalize
  k_attn<<<gconv, 32, 0, stream>>>(qb, kcp, vcr, gg, den, unit);

  // output projection
  k_gemm_out<<<gproj, 32, 0, stream>>>(unit, wpo, wo_b, out);
}
